// DFMSDA_43327630082578
// MI455X (gfx1250) — matrix operand order repaired and checked by reading, not compiled, not run on hardware
//
#include <hip/hip_runtime.h>
#include <hip/hip_bf16.h>
#include <math.h>

// ---------------- types ----------------
typedef __bf16 bf16_t;
typedef __attribute__((ext_vector_type(16))) __bf16 v16bf;
typedef __attribute__((ext_vector_type(8)))  __bf16 v8bf;
typedef __attribute__((ext_vector_type(8)))  float  v8f;

// ---------------- problem constants ----------------
static constexpr int DIMC   = 256;
static constexpr int Bn     = 4;
static constexpr int Hn     = 64;
static constexpr int Wn     = 64;
static constexpr int HWc    = Hn * Wn;        // 4096
static constexpr int Mrows  = Bn * HWc;       // 16384
static constexpr int HIDDENc= 4 * DIMC;       // 1024
static constexpr float SCALE = 0.17677669529663689f; // 32^-0.5

static constexpr int KCHUNK = 256;            // K panel staged in LDS
static constexpr int LDSPAD = 264;            // 256 + 8 bf16 pad (bank-conflict-free)

// ---------------- f32 -> bf16 convert ----------------
__global__ __launch_bounds__(256) void cvt_bf16_kernel(const float* __restrict__ in,
                                                       bf16_t* __restrict__ out, int n) {
    int i = blockIdx.x * 256 + threadIdx.x;
    if (i < n) out[i] = (bf16_t)in[i];
}

// ---------------- prep: x1 = xa - xb; LN(x1), LN(xb) ----------------
// x layout: (B, C, H, W); one thread per spatial position (coalesced reads).
__global__ __launch_bounds__(256) void prep_ln_kernel(
    const float* __restrict__ xa, const float* __restrict__ xb,
    const float* __restrict__ g,  const float* __restrict__ be,
    float* __restrict__ x1res, bf16_t* __restrict__ x1n, bf16_t* __restrict__ x2n) {
    int m  = blockIdx.x * 256 + threadIdx.x;      // 0..16383
    int b  = m >> 12;
    int sp = m & (HWc - 1);
    const float* pa = xa + (size_t)b * DIMC * HWc + sp;
    const float* pb = xb + (size_t)b * DIMC * HWc + sp;
    float s1 = 0.f, q1 = 0.f, s2 = 0.f, q2 = 0.f;
    for (int c = 0; c < DIMC; ++c) {
        float va = pa[(size_t)c * HWc], vb = pb[(size_t)c * HWc];
        float v1 = va - vb;
        s1 += v1; q1 += v1 * v1; s2 += vb; q2 += vb * vb;
    }
    const float inv = 1.0f / (float)DIMC;
    float mu1 = s1 * inv, mu2 = s2 * inv;
    float rs1 = rsqrtf(q1 * inv - mu1 * mu1 + 1e-5f);
    float rs2 = rsqrtf(q2 * inv - mu2 * mu2 + 1e-5f);
    for (int c = 0; c < DIMC; ++c) {
        float va = pa[(size_t)c * HWc], vb = pb[(size_t)c * HWc];
        float v1 = va - vb;
        size_t o = (size_t)m * DIMC + c;
        x1res[o] = v1;
        x1n[o] = (bf16_t)((v1 - mu1) * rs1 * g[c] + be[c]);
        x2n[o] = (bf16_t)((vb - mu2) * rs2 * g[c] + be[c]);
    }
}

// ---------------- LN over rows of [M,256] f32 -> bf16 ----------------
__global__ __launch_bounds__(256) void ln_rows_kernel(
    const float* __restrict__ xin, const float* __restrict__ g,
    const float* __restrict__ be, bf16_t* __restrict__ out) {
    int row  = blockIdx.x * 8 + (threadIdx.x >> 5);
    int lane = threadIdx.x & 31;
    const float* p = xin + (size_t)row * DIMC + lane * 8;
    float v[8];
    float s = 0.f, q = 0.f;
#pragma unroll
    for (int i = 0; i < 8; ++i) { v[i] = p[i]; s += v[i]; q += v[i] * v[i]; }
#pragma unroll
    for (int off = 16; off > 0; off >>= 1) {
        s += __shfl_xor(s, off, 32);
        q += __shfl_xor(q, off, 32);
    }
    const float inv = 1.0f / (float)DIMC;
    float mu = s * inv;
    float rs = rsqrtf(q * inv - mu * mu + 1e-5f);
    int c0 = lane * 8;
#pragma unroll
    for (int i = 0; i < 8; ++i)
        out[(size_t)row * DIMC + c0 + i] = (bf16_t)((v[i] - mu) * rs * g[c0 + i] + be[c0 + i]);
}

// ---------------- WMMA fragment loads (layouts per ISA 7.12.2) ----------------
__device__ inline v16bf load_a_frag(const bf16_t* __restrict__ A, int lda,
                                    int m0, int k0, int lane) {
    int row = m0 + (lane & 15);
    int kb  = (lane >> 4) * 8;              // lanes 0-15: K 0..7,16..23 ; 16-31: 8..15,24..31
    const bf16_t* p = A + (size_t)row * lda + k0 + kb;
    v8bf lo = *(const v8bf*)(p);
    v8bf hi = *(const v8bf*)(p + 16);
    v16bf a;
#pragma unroll
    for (int i = 0; i < 8; ++i) { a[i] = lo[i]; a[i + 8] = hi[i]; }
    return a;
}

// B fragment from LDS-staged panel bs[64][LDSPAD] (ko = K offset within chunk)
__device__ inline v16bf load_b_frag_lds(const bf16_t* bs, int nt, int ko, int lane) {
    int loc = nt * 16 + (lane & 15);
    int kb  = (lane >> 4) * 8;              // lanes 0-15: K 0..7,16..23 ; 16-31: 8..15,24..31
    const bf16_t* p = bs + loc * LDSPAD + ko + kb;
    v8bf lo = *(const v8bf*)(p);
    v8bf hi = *(const v8bf*)(p + 16);
    v16bf b;
#pragma unroll
    for (int i = 0; i < 8; ++i) { b[i] = lo[i]; b[i + 8] = hi[i]; }
    return b;
}

// ---------------- WMMA GEMM: out[m,n] = sum_k A[m,k] * W[n,k] (+epilogue) ----
// Block = 8 waves; each wave computes 32 rows x 64 cols (2 M-tiles x 4 N-tiles).
// The block's 64-column weight panel is staged in LDS via async global->LDS
// copies (ASYNCcnt) in K-chunks of 256, shared by all 8 waves.
// MODE 0: bf16 plain           MODE 1: f32 = acc + bias + res
// MODE 2: bf16 gelu(acc+bias)  MODE 3: f32 scatter BCHW = acc + bias + res
template <int MODE>
__global__ __launch_bounds__(256) void gemm_wmma_kernel(
    const bf16_t* __restrict__ A, const bf16_t* __restrict__ W,
    const float* __restrict__ bias, const float* __restrict__ res,
    float* __restrict__ outf, bf16_t* __restrict__ outb,
    int N, int K) {
    __shared__ bf16_t bs[64 * LDSPAD];            // 33792 B

    const int lane = threadIdx.x & 31;
    const int wave = threadIdx.x >> 5;
    const int m0 = blockIdx.y * 256 + wave * 32;  // wave's 32 output rows
    const int n0 = blockIdx.x * 64;               // block's 64 output cols

    v8f acc[2][4];
#pragma unroll
    for (int mt = 0; mt < 2; ++mt)
#pragma unroll
        for (int nt = 0; nt < 4; ++nt) {
            v8f z = {0.f,0.f,0.f,0.f,0.f,0.f,0.f,0.f};
            acc[mt][nt] = z;
        }

    for (int kc = 0; kc < K; kc += KCHUNK) {
        // ---- stage 64 x 256 bf16 weight panel into LDS (async, 16B units) ----
        // chunk c (0..2047): row j = c>>5, k-offset = (c&31)*8 elements
        {
            int c = threadIdx.x;                   // 8 iterations of 256 threads
#pragma unroll
            for (int it = 0; it < 8; ++it, c += 256) {
                int j  = c >> 5;
                int ke = (c & 31) * 8;
                unsigned lds_off = (unsigned)(uintptr_t)(&bs[j * LDSPAD + ke]);
                const bf16_t* gsrc = W + (size_t)(n0 + j) * K + kc + ke;
                asm volatile("global_load_async_to_lds_b128 %0, %1, off"
                             :: "v"(lds_off), "v"(gsrc) : "memory");
            }
            asm volatile("s_wait_asynccnt 0" ::: "memory");
        }
        __syncthreads();

        // ---- K inner loop over the staged chunk ----
        for (int ko = 0; ko < KCHUNK; ko += 32) {
            int k0 = kc + ko;
            v16bf a0 = load_a_frag(A, K, m0,      k0, lane);
            v16bf a1 = load_a_frag(A, K, m0 + 16, k0, lane);
#pragma unroll
            for (int nt = 0; nt < 4; ++nt) {
                v16bf b = load_b_frag_lds(bs, nt, ko, lane);
                acc[0][nt] = __builtin_amdgcn_wmma_f32_16x16x32_bf16(
                    false, a0, false, b, (short)0, acc[0][nt], false, false);
                acc[1][nt] = __builtin_amdgcn_wmma_f32_16x16x32_bf16(
                    false, a1, false, b, (short)0, acc[1][nt], false, false);
            }
        }
        __syncthreads();                           // before next chunk overwrites LDS
    }

    // ---- epilogue ----
#pragma unroll
    for (int mt = 0; mt < 2; ++mt) {
#pragma unroll
        for (int nt = 0; nt < 4; ++nt) {
            int n = n0 + 16 * nt + (lane & 15);
#pragma unroll
            for (int r = 0; r < 8; ++r) {
                int m = m0 + mt * 16 + r + ((lane >> 4) << 3);
                float v = acc[mt][nt][r];
                size_t o = (size_t)m * N + n;
                if (MODE == 0) {
                    outb[o] = (bf16_t)v;
                } else if (MODE == 1) {
                    outf[o] = v + bias[n] + res[o];
                } else if (MODE == 2) {
                    float z = v + bias[n];
                    outb[o] = (bf16_t)(0.5f * z * (1.0f + erff(z * 0.70710678118654752f)));
                } else {
                    float z = v + bias[n] + res[o];
                    int b_ = m >> 12, sp = m & (HWc - 1);
                    outf[(size_t)b_ * N * HWc + (size_t)n * HWc + sp] = z;
                }
            }
        }
    }
}

// ---------------- dilated local attention (VALU, wave per (pos, head)) ------
__global__ __launch_bounds__(256) void attn_kernel(
    const bf16_t* __restrict__ Q, const bf16_t* __restrict__ K,
    const bf16_t* __restrict__ V, bf16_t* __restrict__ O) {
    int wid  = blockIdx.x * 8 + (threadIdx.x >> 5);
    int lane = threadIdx.x & 31;
    int pos  = wid >> 3;           // 0..16383
    int hg   = wid & 7;            // global head 0..7
    int dil  = (hg >> 1) + 1;      // DILATIONS = (1,2,3,4), 2 heads per dilation
    int b    = pos >> 12;
    int sp   = pos & (HWc - 1);
    int hh   = sp >> 6;
    int ww   = sp & 63;
    int ch   = hg * 32 + lane;

    float qv = (float)Q[(size_t)pos * DIMC + ch];

    float logit[9];
    int   nbr[9];
    bool  ok[9];
#pragma unroll
    for (int ki = 0; ki < 3; ++ki) {
#pragma unroll
        for (int kj = 0; kj < 3; ++kj) {
            int j = ki * 3 + kj;
            int y = hh + (ki - 1) * dil;
            int x = ww + (kj - 1) * dil;
            ok[j]  = (y >= 0) && (y < Hn) && (x >= 0) && (x < Wn);
            nbr[j] = (b << 12) + y * Wn + x;
            float part = ok[j] ? qv * (float)K[(size_t)nbr[j] * DIMC + ch] : 0.f;
#pragma unroll
            for (int off = 16; off > 0; off >>= 1) part += __shfl_xor(part, off, 32);
            logit[j] = part * SCALE;   // zero-padded taps yield exact 0 logits
        }
    }
    float mx = logit[0];
#pragma unroll
    for (int j = 1; j < 9; ++j) mx = fmaxf(mx, logit[j]);
    float den = 0.f, w[9];
#pragma unroll
    for (int j = 0; j < 9; ++j) { w[j] = __expf(logit[j] - mx); den += w[j]; }
    float rden = 1.0f / den;
    float acc = 0.f;
#pragma unroll
    for (int j = 0; j < 9; ++j)
        if (ok[j]) acc += w[j] * rden * (float)V[(size_t)nbr[j] * DIMC + ch];
    O[(size_t)pos * DIMC + ch] = (bf16_t)acc;
}

// ---------------- host launch ----------------
extern "C" void kernel_launch(void* const* d_in, const int* in_sizes, int n_in,
                              void* d_out, int out_size, void* d_ws, size_t ws_size,
                              hipStream_t stream) {
    const float* x       = (const float*)d_in[0];
    const float* norm1_w = (const float*)d_in[1];
    const float* norm1_b = (const float*)d_in[2];
    const float* q_w     = (const float*)d_in[3];
    const float* k_w     = (const float*)d_in[4];
    const float* v_w     = (const float*)d_in[5];
    const float* proj_w  = (const float*)d_in[6];
    const float* proj_b  = (const float*)d_in[7];
    const float* norm2_w = (const float*)d_in[8];
    const float* norm2_b = (const float*)d_in[9];
    const float* fc1_w   = (const float*)d_in[10];
    const float* fc1_b   = (const float*)d_in[11];
    const float* fc2_w   = (const float*)d_in[12];
    const float* fc2_b   = (const float*)d_in[13];
    float* outp = (float*)d_out;

    char* ws = (char*)d_ws;
    // workspace layout (all offsets 256B-aligned by construction)
    bf16_t* wq    = (bf16_t*)(ws + 0);           // 2*256*256
    bf16_t* wk    = (bf16_t*)(ws + 262144);
    bf16_t* wv    = (bf16_t*)(ws + 524288);
    bf16_t* wp    = (bf16_t*)(ws + 786432);
    bf16_t* wfc1  = (bf16_t*)(ws + 1048576);     // 2*1024*256
    bf16_t* wfc2  = (bf16_t*)(ws + 2097152);     // 2*256*1024
    float*  x1res = (float*) (ws + 3145728);     // M*256 f32
    bf16_t* x1n   = (bf16_t*)(ws + 19922944);    // M*256 bf16
    bf16_t* x2n   = (bf16_t*)(ws + 28311552);
    bf16_t* Qb    = (bf16_t*)(ws + 36700160);
    bf16_t* Kb    = (bf16_t*)(ws + 45088768);
    bf16_t* Vb    = (bf16_t*)(ws + 53477376);
    bf16_t* attnb = (bf16_t*)(ws + 61865984);
    float*  xaft  = (float*) (ws + 70254592);    // M*256 f32
    bf16_t* hb    = (bf16_t*)(ws + 87031808);    // M*256 bf16
    bf16_t* h1b   = (bf16_t*)(ws + 95420416);    // M*1024 bf16

    // one-time weight conversion (all blocks at once)
    auto cvt = [&](const float* src, bf16_t* dst, int n) {
        cvt_bf16_kernel<<<(n + 255) / 256, 256, 0, stream>>>(src, dst, n);
    };
    cvt(q_w,    wq,   2 * DIMC * DIMC);
    cvt(k_w,    wk,   2 * DIMC * DIMC);
    cvt(v_w,    wv,   2 * DIMC * DIMC);
    cvt(proj_w, wp,   2 * DIMC * DIMC);
    cvt(fc1_w,  wfc1, 2 * HIDDENc * DIMC);
    cvt(fc2_w,  wfc2, 2 * DIMC * HIDDENc);

    const size_t tsz = (size_t)Bn * DIMC * HWc;  // elements per input tensor slice

    for (int bi = 0; bi < 2; ++bi) {
        const float* xa = x + (bi == 0 ? 0 : tsz);   // minuend
        const float* xb = x + (bi == 0 ? tsz : 0);   // subtrahend & x2/x3 source

        prep_ln_kernel<<<Mrows / 256, 256, 0, stream>>>(
            xa, xb, norm1_w + bi * DIMC, norm1_b + bi * DIMC, x1res, x1n, x2n);

        dim3 g256(DIMC / 64, Mrows / 256);           // (4, 64)
        gemm_wmma_kernel<0><<<g256, 256, 0, stream>>>(
            x1n, wq + (size_t)bi * DIMC * DIMC, nullptr, nullptr, nullptr, Qb, DIMC, DIMC);
        gemm_wmma_kernel<0><<<g256, 256, 0, stream>>>(
            x2n, wk + (size_t)bi * DIMC * DIMC, nullptr, nullptr, nullptr, Kb, DIMC, DIMC);
        gemm_wmma_kernel<0><<<g256, 256, 0, stream>>>(
            x2n, wv + (size_t)bi * DIMC * DIMC, nullptr, nullptr, nullptr, Vb, DIMC, DIMC);

        attn_kernel<<<Mrows, 256, 0, stream>>>(Qb, Kb, Vb, attnb);

        gemm_wmma_kernel<1><<<g256, 256, 0, stream>>>(
            attnb, wp + (size_t)bi * DIMC * DIMC, proj_b + bi * DIMC, x1res, xaft, nullptr,
            DIMC, DIMC);

        ln_rows_kernel<<<Mrows / 8, 256, 0, stream>>>(
            xaft, norm2_w + bi * DIMC, norm2_b + bi * DIMC, hb);

        dim3 g1024(HIDDENc / 64, Mrows / 256);       // (16, 64)
        gemm_wmma_kernel<2><<<g1024, 256, 0, stream>>>(
            hb, wfc1 + (size_t)bi * HIDDENc * DIMC, fc1_b + bi * HIDDENc, nullptr, nullptr, h1b,
            HIDDENc, DIMC);

        gemm_wmma_kernel<3><<<g256, 256, 0, stream>>>(
            h1b, wfc2 + (size_t)bi * DIMC * HIDDENc, fc2_b + bi * DIMC, xaft,
            outp + (size_t)bi * tsz, nullptr, DIMC, HIDDENc);
    }
}
